// PersonaGRU_50474455663215
// MI455X (gfx1250) — compile-verified, matrix operand order checked by reading, not run on hardware
//
#include <hip/hip_runtime.h>
#include <hip/hip_bf16.h>

// ---------------------------------------------------------------------------
// Bidirectional masked GRU, fused, bf16 WMMA (gfx1250 / CDNA5, wave32).
//   B=4096, S=10, H=512, D_IN=1024.  out: (B, S, 4H) f32.
// Compute-bound (~773 GFLOP vs ~520MB min HBM traffic => HBM floor ~22us at
// 23.3TB/s; bf16 WMMA compute dominates). Fully fused recurrence:
//   - block = 32 batch rows x 1 direction (grid 128x2), 256 threads (8 waves)
//   - h kept in LDS bf16, ping-pong buffers (WAR-safe across timesteps)
//   - weights pre-converted to bf16 (9.4MB) -> resident in 192MB L2; every
//     block re-streams them per timestep; WGP-scope prefetch pulls the next
//     K-tile into WGP$ ahead of the clause'd global_load_b128 pairs
//   - per wave job: 16x16 column tile, 12 f32 accumulators (3 gates x 2
//     M-tiles x gx/gh); A-fragment shared across 3 gates, B-fragment shared
//     across 2 M-tiles; gate nonlinearity fused in registers.
// ---------------------------------------------------------------------------

#define B_    4096
#define S_    10
#define H_    512
#define DIN_  1024
#define G3_   1536
#define ROWS_ 32

typedef __attribute__((ext_vector_type(16))) __bf16 v16bf;
typedef __attribute__((ext_vector_type(8)))  float  v8f;

union BFrag {
  uint4 u[2];
  v16bf v;
};

__device__ __forceinline__ unsigned short f2bf(float f) {
  unsigned int u = __float_as_uint(f);
  u += 0x7FFFu + ((u >> 16) & 1u);          // round-to-nearest-even
  return (unsigned short)(u >> 16);
}
__device__ __forceinline__ float bf2f(unsigned short h) {
  return __uint_as_float(((unsigned int)h) << 16);
}
__device__ __forceinline__ float sigf(float x) {
  return 1.0f / (1.0f + __expf(-x));
}
__device__ __forceinline__ float tanhfast(float x) {
  return 2.0f / (1.0f + __expf(-2.0f * x)) - 1.0f;
}
__device__ __forceinline__ v8f wmma_bf16(const BFrag& a, const BFrag& b, v8f c) {
  // D = A(16x32 bf16) x B(32x16 bf16) + C(16x16 f32)
  return __builtin_amdgcn_wmma_f32_16x16x32_bf16(false, a.v, false, b.v,
                                                 (short)0, c, false, false);
}

// ----------------------------- weight convert ------------------------------
__global__ void persona_conv_weights(const float* __restrict__ wif,
                                     const float* __restrict__ wib,
                                     const float* __restrict__ whf,
                                     const float* __restrict__ whb,
                                     unsigned short* __restrict__ wih_bf,
                                     unsigned short* __restrict__ whh_bf) {
  const int NI = G3_ * DIN_;
  const int NH = G3_ * H_;
  int i = blockIdx.x * 256 + threadIdx.x;
  if (i < NI) {
    wih_bf[i]      = f2bf(wif[i]);
    wih_bf[NI + i] = f2bf(wib[i]);
  }
  if (i < NH) {
    whh_bf[i]      = f2bf(whf[i]);
    whh_bf[NH + i] = f2bf(whb[i]);
  }
}

// ------------------------------- fused GRU ---------------------------------
__global__ __launch_bounds__(256, 1)
void persona_gru_kernel(const float* __restrict__ x,
                        const int*   __restrict__ mask_fr,
                        const int*   __restrict__ mask_bk,
                        const float* __restrict__ hx,
                        const float* __restrict__ bih_f,
                        const float* __restrict__ bhh_f,
                        const float* __restrict__ bih_b,
                        const float* __restrict__ bhh_b,
                        const unsigned short* __restrict__ wih_bf,
                        const unsigned short* __restrict__ whh_bf,
                        float* __restrict__ out) {
  // Dynamic LDS: x tile 32x1024 bf16 (64KB) + h ping-pong 2x2x32x512 bf16 (128KB)
  extern __shared__ unsigned short smem[];
  unsigned short* x_bf = smem;                       // [ROWS][DIN]
  unsigned short* h_bf = smem + ROWS_ * DIN_;        // [buf2][cell2][ROWS][H]

  const int tid  = threadIdx.x;
  const int lane = tid & 31;
  const int wave = tid >> 5;
  const int dir  = blockIdx.y;                       // 0 = fwd scan, 1 = bwd scan
  const int b0   = blockIdx.x * ROWS_;
  const int* mask = dir ? mask_bk : mask_fr;

  // init h = hx (broadcast initial state)
  for (int i = tid; i < 2 * ROWS_ * H_; i += 256) {
    int cell = i >> 14;                              // / (ROWS_*H_)
    int col  = i & (H_ - 1);
    h_bf[i] = f2bf(hx[cell * H_ + col]);
  }
  __syncthreads();

  // lane decomposition for documented 16-bit WMMA fragment layouts
  const int arow0 = lane & 15;                       // A-frag row, M-tile 0
  const int arow1 = 16 + (lane & 15);                // A-frag row, M-tile 1
  const int klsel = (lane & 16) ? 8 : 0;             // lanes>=16 take K+8 / K+24
  const int crow  = ((lane >> 4) & 1) << 3;          // C/D: lanes>=16 hold M+8
  const int ccol  = lane & 15;                       // C/D + B column index

  for (int t = 0; t < S_; ++t) {
    const int cur = t & 1;
    const int nxt = cur ^ 1;
    const int tx  = dir ? (S_ - 1 - t) : t;          // input time index
    const int so  = tx;                              // output time index (== tx both dirs)

    // ---- stage x tile: f32 -> bf16 into LDS (pairs, b64 load / b32 store) ----
    unsigned int* x_bf32 = (unsigned int*)x_bf;
    for (int p = tid; p < ROWS_ * DIN_ / 2; p += 256) {
      int row = p >> 9;                              // 512 pairs per row
      int kp  = (p & 511) * 2;
      const float2 v = *(const float2*)(x + ((size_t)(b0 + row) * S_ + tx) * DIN_ + kp);
      x_bf32[p] = (unsigned int)f2bf(v.x) | ((unsigned int)f2bf(v.y) << 16);
    }

    // ---- mask reset: h <- hx where mask[b,t]==0 (in place on current buffer) ----
    unsigned short* hc = h_bf + cur * (2 * ROWS_ * H_);
    for (int i = tid; i < 2 * ROWS_ * H_; i += 256) {
      int cell = i >> 14;
      int row  = (i >> 9) & (ROWS_ - 1);
      int col  = i & (H_ - 1);
      if (mask[(size_t)(b0 + row) * S_ + t] == 0)
        hc[i] = f2bf(hx[cell * H_ + col]);
    }
    __syncthreads();

    // ---- 64 jobs = {cell 0|1} x {32 column tiles of 16}; 8 per wave ----
    for (int jj = 0; jj < 8; ++jj) {
      const int job  = wave * 8 + jj;
      const int cell = job >> 5;
      const int j0   = (job & 31) << 4;
      const int ncol = j0 + ccol;

      const unsigned short* wih_c = wih_bf + (size_t)cell * G3_ * DIN_;
      const unsigned short* whh_c = whh_bf + (size_t)cell * G3_ * H_;
      const float* bih = cell ? bih_b : bih_f;
      const float* bhh = cell ? bhh_b : bhh_f;
      const unsigned short* hcell = hc + cell * (ROWS_ * H_);

      // accumulators: [gate r/z/n][M-tile], gx and gh kept separate (n-gate needs r*hn)
      v8f ax[3][2], ah[3][2];
#pragma unroll
      for (int g = 0; g < 3; ++g) {
        const float bi = bih[g * H_ + ncol];
        const float bh = bhh[g * H_ + ncol];
        const v8f vi = {bi, bi, bi, bi, bi, bi, bi, bi};
        const v8f vh = {bh, bh, bh, bh, bh, bh, bh, bh};
        ax[g][0] = vi; ax[g][1] = vi;
        ah[g][0] = vh; ah[g][1] = vh;
      }

      // gx: x(32x1024) @ w_ih^T -> 32 K-steps of 16x16x32 WMMA, 3 gates x 2 M-tiles
      for (int kk = 0; kk < DIN_; kk += 32) {
        const int k0 = kk + klsel;
        BFrag a0, a1;
        a0.u[0] = *(const uint4*)&x_bf[arow0 * DIN_ + k0];
        a0.u[1] = *(const uint4*)&x_bf[arow0 * DIN_ + k0 + 16];
        a1.u[0] = *(const uint4*)&x_bf[arow1 * DIN_ + k0];
        a1.u[1] = *(const uint4*)&x_bf[arow1 * DIN_ + k0 + 16];
#pragma unroll
        for (int g = 0; g < 3; ++g) {
          const unsigned short* wp = wih_c + (size_t)(g * H_ + ncol) * DIN_ + k0;
          // locality=3 -> WGP-scope prefetch: fills WGP$ (SYS scope would skip it)
          __builtin_prefetch(wp + 32, 0, 3);
          BFrag bw;
          bw.u[0] = *(const uint4*)wp;
          bw.u[1] = *(const uint4*)(wp + 16);
          ax[g][0] = wmma_bf16(a0, bw, ax[g][0]);
          ax[g][1] = wmma_bf16(a1, bw, ax[g][1]);
        }
      }
      // gh: h(32x512) @ w_hh^T -> 16 K-steps (the true recurrent dependency)
      for (int kk = 0; kk < H_; kk += 32) {
        const int k0 = kk + klsel;
        BFrag a0, a1;
        a0.u[0] = *(const uint4*)&hcell[arow0 * H_ + k0];
        a0.u[1] = *(const uint4*)&hcell[arow0 * H_ + k0 + 16];
        a1.u[0] = *(const uint4*)&hcell[arow1 * H_ + k0];
        a1.u[1] = *(const uint4*)&hcell[arow1 * H_ + k0 + 16];
#pragma unroll
        for (int g = 0; g < 3; ++g) {
          const unsigned short* wp = whh_c + (size_t)(g * H_ + ncol) * H_ + k0;
          __builtin_prefetch(wp + 32, 0, 3);         // WGP-scope prefetch, next K tile
          BFrag bw;
          bw.u[0] = *(const uint4*)wp;
          bw.u[1] = *(const uint4*)(wp + 16);
          ah[g][0] = wmma_bf16(a0, bw, ah[g][0]);
          ah[g][1] = wmma_bf16(a1, bw, ah[g][1]);
        }
      }

      // fused gate math + writes (h_old reads precede the barrier; h_new -> other buffer)
      unsigned short* hnext = h_bf + nxt * (2 * ROWS_ * H_) + cell * (ROWS_ * H_);
#pragma unroll
      for (int mt = 0; mt < 2; ++mt) {
#pragma unroll
        for (int i = 0; i < 8; ++i) {
          const int   row  = mt * 16 + crow + i;     // C/D layout: M = vgpr + 8*(lane>=16)
          const float hold = bf2f(hcell[row * H_ + ncol]);
          const float r    = sigf(ax[0][mt][i] + ah[0][mt][i]);
          const float z    = sigf(ax[1][mt][i] + ah[1][mt][i]);
          const float n    = tanhfast(ax[2][mt][i] + r * ah[2][mt][i]);
          const float hnew = (1.0f - z) * n + z * hold;
          hnext[row * H_ + ncol] = f2bf(hnew);
          out[((size_t)(b0 + row) * S_ + so) * (4 * H_) +
              (size_t)dir * (2 * H_) + cell * H_ + ncol] = hnew;
        }
      }
    }
    __syncthreads();
  }
}

// ------------------------------- launcher ----------------------------------
extern "C" void kernel_launch(void* const* d_in, const int* in_sizes, int n_in,
                              void* d_out, int out_size, void* d_ws, size_t ws_size,
                              hipStream_t stream) {
  (void)in_sizes; (void)n_in; (void)out_size; (void)ws_size;

  const float* x   = (const float*)d_in[0];
  const int*   mfr = (const int*)d_in[1];
  const int*   mbk = (const int*)d_in[2];
  const float* hx  = (const float*)d_in[3];
  const float* wif = (const float*)d_in[4];
  const float* whf = (const float*)d_in[5];
  const float* bif = (const float*)d_in[6];
  const float* bhf = (const float*)d_in[7];
  const float* wib = (const float*)d_in[8];
  const float* whb = (const float*)d_in[9];
  const float* bib = (const float*)d_in[10];
  const float* bhb = (const float*)d_in[11];
  float* out = (float*)d_out;

  // workspace: bf16 weights, [cell][n][k] row-major (9.4 MB total, L2-resident)
  unsigned short* wih_bf = (unsigned short*)d_ws;
  unsigned short* whh_bf = wih_bf + (size_t)2 * G3_ * DIN_;

  persona_conv_weights<<<(G3_ * DIN_ + 255) / 256, 256, 0, stream>>>(
      wif, wib, whf, whb, wih_bf, whh_bf);

  const int smem_bytes = (ROWS_ * DIN_ + 2 * 2 * ROWS_ * H_) * (int)sizeof(unsigned short); // 192KB
  hipFuncSetAttribute((const void*)persona_gru_kernel,
                      hipFuncAttributeMaxDynamicSharedMemorySize, smem_bytes);

  persona_gru_kernel<<<dim3(B_ / ROWS_, 2), 256, smem_bytes, stream>>>(
      x, mfr, mbk, hx, bif, bhf, bib, bhb, wih_bf, whh_bf, out);
}
